// PUGeo_40578851012636
// MI455X (gfx1250) — matrix operand order repaired and checked by reading, not compiled, not run on hardware
//
#include <hip/hip_runtime.h>

typedef __attribute__((ext_vector_type(16))) _Float16 v16h;
typedef __attribute__((ext_vector_type(8)))  _Float16 v8h;
typedef __attribute__((ext_vector_type(8)))  float    v8f;

#define KNN 20
#define FD 128
#define LDF 1152
#define NPTS 16384          /* B*N = 2*8192 */
#define NCHUNK 64           /* NPTS/256     */
#define ROWS_E (NPTS*KNN)   /* 327680 edge rows */

union HV { v16h v; v8h h[2]; };

// ---------------------------------------------------------------------------
// Fold conv weights to f16 (padded) + BN into scale/shift.
__global__ void prep_layer(const float* __restrict__ W, const float* __restrict__ b,
                           const float* __restrict__ g, const float* __restrict__ be,
                           const float* __restrict__ m, const float* __restrict__ v,
                           int Cout, int Cin, int CoutP, int KP,
                           _Float16* __restrict__ Wh, float* __restrict__ scale,
                           float* __restrict__ shift, int hasBN)
{
    int tid = blockIdx.x * blockDim.x + threadIdx.x;
    if (tid >= CoutP * KP) return;
    int co = tid / KP, ci = tid % KP;
    float w = (co < Cout && ci < Cin) ? W[co * Cin + ci] : 0.f;
    Wh[co * KP + ci] = (_Float16)w;
    if (ci == 0) {
        float s = 0.f, sh = 0.f;
        if (co < Cout) {
            if (hasBN) { s = g[co] * rsqrtf(v[co] + 1e-5f); sh = (b[co] - m[co]) * s + be[co]; }
            else       { s = 1.f; sh = b[co]; }
        }
        scale[co] = s; shift[co] = sh;
    }
}

// x (B,3,N) -> xt (B*N, 3)
__global__ void transpose_x(const float* __restrict__ x, float* __restrict__ xt)
{
    int tid = blockIdx.x * blockDim.x + threadIdx.x;
    if (tid >= 2 * 3 * 8192) return;
    int b = tid / (3 * 8192), r = tid % (3 * 8192);
    int ch = r / 8192, n = r % 8192;
    xt[((size_t)b * 8192 + n) * 3 + ch] = x[tid];
}

// ---------------------------------------------------------------------------
// KNN round 1: 3-D coordinates, one block per 256-pt chunk.
__global__ void knn3_kernel(const float* __restrict__ xt, int* __restrict__ idx)
{
    __shared__ float px[256], py[256], pz[256];
    int c = blockIdx.x, i = threadIdx.x;
    int base = c * 256;
    px[i] = xt[(size_t)(base + i) * 3 + 0];
    py[i] = xt[(size_t)(base + i) * 3 + 1];
    pz[i] = xt[(size_t)(base + i) * 3 + 2];
    __syncthreads();
    float bd[KNN]; int bi[KNN];
#pragma unroll
    for (int t = 0; t < KNN; ++t) { bd[t] = 3.4e38f; bi[t] = 0; }
    float xi = px[i], yi = py[i], zi = pz[i];
    for (int j = 0; j < 256; ++j) {
        float dx = px[j] - xi, dy = py[j] - yi, dz = pz[j] - zi;
        float d = dx * dx + dy * dy + dz * dz;
        if (d < bd[KNN - 1]) {
            bd[KNN - 1] = d; bi[KNN - 1] = j;
#pragma unroll
            for (int t = KNN - 1; t > 0; --t)
                if (bd[t] < bd[t - 1]) {
                    float td = bd[t]; bd[t] = bd[t - 1]; bd[t - 1] = td;
                    int   ti = bi[t]; bi[t] = bi[t - 1]; bi[t - 1] = ti;
                }
        }
    }
#pragma unroll
    for (int t = 0; t < KNN; ++t) idx[(size_t)(base + i) * KNN + t] = bi[t];
}

// ---------------------------------------------------------------------------
// Gram matrix G = X * X^T per chunk via WMMA (for 128-D KNN distance matrix).
__global__ void gram_wmma(const _Float16* __restrict__ feat, int colOff, float* __restrict__ G)
{
    int wave = (blockIdx.x * blockDim.x + threadIdx.x) >> 5;
    int lane = threadIdx.x & 31;
    int chunk = wave >> 8;
    int tt = wave & 255;
    int tm = tt >> 4, tn = tt & 15;
    int g = lane >> 4, mr = lane & 15;
    const _Float16* X = feat + (size_t)chunk * 256 * LDF + colOff;
    const _Float16* arow = X + (size_t)(tm * 16 + mr) * LDF;
    const _Float16* brow = X + (size_t)(tn * 16 + mr) * LDF;
    v8f acc = {};
    for (int kk = 0; kk < FD; kk += 32) {
        HV a, b;
        a.h[0] = *(const v8h*)(arow + kk + g * 8);
        a.h[1] = *(const v8h*)(arow + kk + 16 + g * 8);
        b.h[0] = *(const v8h*)(brow + kk + g * 8);
        b.h[1] = *(const v8h*)(brow + kk + 16 + g * 8);
        acc = __builtin_amdgcn_wmma_f32_16x16x32_f16(false, a.v, false, b.v,
                                                     (short)0, acc, false, false);
    }
    float* Gc = G + (size_t)chunk * 65536;
#pragma unroll
    for (int i = 0; i < 8; ++i)
        Gc[(size_t)(tm * 16 + g * 8 + i) * 256 + tn * 16 + mr] = acc[i];
}

// top-20 from d_ij = |xi|^2 + |xj|^2 - 2 G_ij ; one block per chunk.
__global__ void knn_feat_topk(const _Float16* __restrict__ feat, int colOff,
                              const float* __restrict__ G, int* __restrict__ idx)
{
    __shared__ float sq[256];
    int chunk = blockIdx.x, i = threadIdx.x;
    int pt = chunk * 256 + i;
    const _Float16* fr = feat + (size_t)pt * LDF + colOff;
    float s = 0.f;
#pragma unroll
    for (int c = 0; c < FD; c += 8) {
        v8h h = *(const v8h*)(fr + c);
#pragma unroll
        for (int e = 0; e < 8; ++e) { float v = (float)h[e]; s += v * v; }
    }
    sq[i] = s;
    __syncthreads();
    const float* Gr = G + (size_t)chunk * 65536 + (size_t)i * 256;
    float bd[KNN]; int bi[KNN];
#pragma unroll
    for (int t = 0; t < KNN; ++t) { bd[t] = 3.4e38f; bi[t] = 0; }
    float si = sq[i];
    for (int j = 0; j < 256; ++j) {
        float d = si + sq[j] - 2.f * Gr[j];
        if (d < bd[KNN - 1]) {
            bd[KNN - 1] = d; bi[KNN - 1] = j;
#pragma unroll
            for (int t = KNN - 1; t > 0; --t)
                if (bd[t] < bd[t - 1]) {
                    float td = bd[t]; bd[t] = bd[t - 1]; bd[t - 1] = td;
                    int   ti = bi[t]; bi[t] = bi[t - 1]; bi[t - 1] = ti;
                }
        }
    }
#pragma unroll
    for (int t = 0; t < KNN; ++t) idx[(size_t)pt * KNN + t] = bi[t];
}

// ---------------------------------------------------------------------------
// Edge gathers: e = [f_nbr - f_ctr , f_ctr]
__global__ void gather_edge3(const float* __restrict__ xt, const int* __restrict__ idx,
                             _Float16* __restrict__ e)
{
    long long tid = (long long)blockIdx.x * blockDim.x + threadIdx.x;
    if (tid >= (long long)ROWS_E * 32) return;
    int c = (int)(tid & 31);
    long long row = tid >> 5;
    int pt = (int)(row / KNN), k = (int)(row % KNN);
    int nbr = (pt & ~255) + idx[(size_t)pt * KNN + k];
    float val = 0.f;
    if (c < 3)      val = xt[(size_t)nbr * 3 + c] - xt[(size_t)pt * 3 + c];
    else if (c < 6) val = xt[(size_t)pt * 3 + (c - 3)];
    e[(size_t)row * 32 + c] = (_Float16)val;
}

__global__ void gather_edge128(const _Float16* __restrict__ feat, int colOff,
                               const int* __restrict__ idx, _Float16* __restrict__ e)
{
    long long tid = (long long)blockIdx.x * blockDim.x + threadIdx.x;
    if (tid >= (long long)ROWS_E * FD) return;
    int c = (int)(tid & 127);
    long long row = tid >> 7;
    int pt = (int)(row / KNN), k = (int)(row % KNN);
    int nbr = (pt & ~255) + idx[(size_t)pt * KNN + k];
    _Float16 fc = feat[(size_t)pt  * LDF + colOff + c];
    _Float16 fn = feat[(size_t)nbr * LDF + colOff + c];
    e[(size_t)row * 256 + c]       = (_Float16)(fn - fc);
    e[(size_t)row * 256 + 128 + c] = fc;
}

// ---------------------------------------------------------------------------
// Fused GEMM, block-cooperative:
//   * one 256-thread block = 8 waves = 8 consecutive M-tiles x ONE group of NT
//     N-tiles (so the whole block shares one weight tile),
//   * weight K-panels (64 wide) staged once per block into LDS (padded stride
//     72 halves = 144B breaks the 256B bank period -> conflict-free b128 reads),
//   * A fragments stream global->VGPR (wave-private rows), reused over NT WMMAs.
// No early returns: every block is full, EXEC all-1s (WMMA requirement).
template<int NT>
__global__ void gemm_wmma_fused(const _Float16* __restrict__ A, int ldA, int aOff,
                                const _Float16* __restrict__ Wh, int KP,
                                const float* __restrict__ scale, const float* __restrict__ shift,
                                _Float16* __restrict__ outh, float* __restrict__ outf,
                                int ldOut, int outOff, int tilesM8, float slope)
{
    constexpr int WROWS = NT * 16;
    constexpr int LSTR  = 64 + 8;                 // padded LDS row stride (halves)
    __shared__ _Float16 wlds[WROWS * LSTR];       // <= 18KB for NT=8
    int gn   = blockIdx.x / tilesM8;
    int tm   = (blockIdx.x % tilesM8) * 8 + (threadIdx.x >> 5);
    int lane = threadIdx.x & 31;
    int g = lane >> 4, mr = lane & 15;
    const _Float16* arow = A  + (size_t)(tm * 16 + mr) * ldA + aOff;
    const _Float16* wsrc = Wh + (size_t)gn * WROWS * KP;
    v8f acc[NT] = {};
    for (int kp0 = 0; kp0 < KP; kp0 += 64) {
        int kplen = (KP - kp0 >= 64) ? 64 : (KP - kp0);   // 64, or 32 for c1
        __syncthreads();
        for (int t = threadIdx.x * 8; t < WROWS * kplen; t += 256 * 8) {
            int row = t / kplen, kw = t % kplen;
            *(v8h*)&wlds[row * LSTR + kw] =
                *(const v8h*)(wsrc + (size_t)row * KP + kp0 + kw);
        }
        __syncthreads();
        for (int kk = 0; kk < kplen; kk += 32) {
            HV a;
            a.h[0] = *(const v8h*)(arow + kp0 + kk + g * 8);
            a.h[1] = *(const v8h*)(arow + kp0 + kk + 16 + g * 8);
            __builtin_prefetch(arow + kp0 + kk + 32, 0, 0);   // global_prefetch_b8
#pragma unroll
            for (int nt = 0; nt < NT; ++nt) {
                const _Float16* wr = &wlds[(nt * 16 + mr) * LSTR + kk + g * 8];
                HV b;
                b.h[0] = *(const v8h*)(wr);
                b.h[1] = *(const v8h*)(wr + 16);
                acc[nt] = __builtin_amdgcn_wmma_f32_16x16x32_f16(false, a.v, false, b.v,
                                                                 (short)0, acc[nt], false, false);
            }
        }
    }
#pragma unroll
    for (int nt = 0; nt < NT; ++nt) {
        int col = (gn * NT + nt) * 16 + mr;
        float sc = scale[col], sh = shift[col];
#pragma unroll
        for (int i = 0; i < 8; ++i) {
            size_t r = (size_t)(tm * 16 + g * 8 + i);
            float y = acc[nt][i] * sc + sh;
            y = y > 0.f ? y : slope * y;
            if (outh) outh[r * ldOut + outOff + col] = (_Float16)y;
            if (outf) outf[r * ldOut + outOff + col] = y;
        }
    }
}

// max + mean pooling over the 20 neighbors -> slots of the 1152-ch feat buffer
__global__ void pool_maxmean(const _Float16* __restrict__ oe, _Float16* __restrict__ feat,
                             int mxOff, int mnOff)
{
    int tid = blockIdx.x * blockDim.x + threadIdx.x;
    if (tid >= NPTS * FD) return;
    int c = tid & 127, pt = tid >> 7;
    float mx = -3.4e38f, s = 0.f;
    for (int k = 0; k < KNN; ++k) {
        float v = (float)oe[((size_t)pt * KNN + k) * FD + c];
        mx = fmaxf(mx, v); s += v;
    }
    feat[(size_t)pt * LDF + mxOff + c] = (_Float16)mx;
    feat[(size_t)pt * LDF + mnOff + c] = (_Float16)(s * (1.f / KNN));
}

// ---------------------------------------------------------------------------
// Quadratic patch expansion + normals.
__global__ void finalize_kernel(const float* __restrict__ xt, const float* __restrict__ coef,
                                const int* __restrict__ sel, const float* __restrict__ uvset,
                                float* __restrict__ out)
{
    int tid = blockIdx.x * blockDim.x + threadIdx.x;
    if (tid >= NPTS * 16) return;
    int r = tid & 15, pt = tid >> 4;
    const float* cf = coef + (size_t)pt * 32;
    const float* uvp = uvset + (size_t)sel[pt] * 32 + r * 2;
    float u = uvp[0], v = uvp[1];
    float uvv[6] = { 1.f, u, v, u * u, u * v, v * v };
    float xyz[3], du[3], dv[3];
#pragma unroll
    for (int t = 0; t < 3; ++t) {
        float a0 = cf[0 + t], a1 = cf[3 + t], a2 = cf[6 + t],
              a3 = cf[9 + t], a4 = cf[12 + t], a5 = cf[15 + t];
        xyz[t] = xt[(size_t)pt * 3 + t] + uvv[0] * a0 + uvv[1] * a1 + uvv[2] * a2
               + uvv[3] * a3 + uvv[4] * a4 + uvv[5] * a5;
        du[t] = a1 + 2.f * u * a3 + v * a4;
        dv[t] = a2 + u * a4 + 2.f * v * a5;
    }
    float n0 = du[1] * dv[2] - du[2] * dv[1];
    float n1 = du[2] * dv[0] - du[0] * dv[2];
    float n2 = du[0] * dv[1] - du[1] * dv[0];
    float len = fmaxf(sqrtf(n0 * n0 + n1 * n1 + n2 * n2), 1e-12f);
    int b = pt >> 13, n = pt & 8191;
    size_t o = ((size_t)b * 131072 + (size_t)n * 16 + r) * 3;
    out[o + 0] = xyz[0]; out[o + 1] = xyz[1]; out[o + 2] = xyz[2];
    float* nrm = out + (size_t)2 * 131072 * 3;
    nrm[o + 0] = n0 / len; nrm[o + 1] = n1 / len; nrm[o + 2] = n2 / len;
}

// ---------------------------------------------------------------------------
extern "C" void kernel_launch(void* const* d_in, const int* in_sizes, int n_in,
                              void* d_out, int out_size, void* d_ws, size_t ws_size,
                              hipStream_t stream)
{
    (void)in_sizes; (void)n_in; (void)out_size; (void)ws_size;
    const float* x     = (const float*)d_in[0];
    const int*   sel   = (const int*)  d_in[1];
    const float* uvset = (const float*)d_in[2];

    char* p = (char*)d_ws;
    auto alloc = [&](size_t bytes) -> void* {
        void* r = (void*)p; p += (bytes + 255) & ~(size_t)255; return r;
    };
    float*     xt   = (float*)    alloc((size_t)NPTS * 3 * 4);
    _Float16*  feat = (_Float16*) alloc((size_t)NPTS * LDF * 2);
    _Float16*  edge = (_Float16*) alloc((size_t)ROWS_E * 256 * 2);
    _Float16*  oeA  = (_Float16*) alloc((size_t)ROWS_E * FD * 2);
    _Float16*  oeB  = (_Float16*) alloc((size_t)ROWS_E * FD * 2);
    float*     G    = (float*)    alloc((size_t)NCHUNK * 65536 * 4);
    int*       idx  = (int*)      alloc((size_t)NPTS * KNN * 4);
    _Float16*  h1b  = (_Float16*) alloc((size_t)NPTS * 256 * 2);
    _Float16*  h2b  = (_Float16*) alloc((size_t)NPTS * 128 * 2);
    _Float16*  h3b  = (_Float16*) alloc((size_t)NPTS * 64 * 2);
    float*     coef = (float*)    alloc((size_t)NPTS * 32 * 4);

    struct Layer { int cout, cin, coutP, kp, hasBN, pidx; _Float16* Wh; float* sc; float* sh; };
    Layer Ls[11] = {
        {128,   6, 128,   32, 1,  3, 0, 0, 0},   // c1
        {128, 128, 128,  128, 1,  9, 0, 0, 0},   // c2
        {128, 256, 128,  256, 1, 15, 0, 0, 0},   // c3
        {128, 256, 128,  256, 1, 21, 0, 0, 0},   // c4
        {128, 256, 128,  256, 1, 27, 0, 0, 0},   // c5
        {128, 256, 128,  256, 1, 33, 0, 0, 0},   // c6
        {128, 256, 128,  256, 1, 39, 0, 0, 0},   // c7
        {256,1152, 256, 1152, 1, 45, 0, 0, 0},   // h1
        {128, 256, 128,  256, 1, 51, 0, 0, 0},   // h2
        { 64, 128,  64,  128, 1, 57, 0, 0, 0},   // h3
        { 18,  64,  32,   64, 0, 63, 0, 0, 0},   // h4 (no BN)
    };
    for (int i = 0; i < 11; ++i) {
        Ls[i].Wh = (_Float16*)alloc((size_t)Ls[i].coutP * Ls[i].kp * 2);
        Ls[i].sc = (float*)   alloc((size_t)Ls[i].coutP * 4);
        Ls[i].sh = (float*)   alloc((size_t)Ls[i].coutP * 4);
        const float* Wp  = (const float*)d_in[Ls[i].pidx];
        const float* bp  = (const float*)d_in[Ls[i].pidx + 1];
        const float* gp  = Ls[i].hasBN ? (const float*)d_in[Ls[i].pidx + 2] : bp;
        const float* bep = Ls[i].hasBN ? (const float*)d_in[Ls[i].pidx + 3] : bp;
        const float* mp  = Ls[i].hasBN ? (const float*)d_in[Ls[i].pidx + 4] : bp;
        const float* vp  = Ls[i].hasBN ? (const float*)d_in[Ls[i].pidx + 5] : bp;
        int tot = Ls[i].coutP * Ls[i].kp;
        prep_layer<<<(tot + 255) / 256, 256, 0, stream>>>(
            Wp, bp, gp, bep, mp, vp, Ls[i].cout, Ls[i].cin,
            Ls[i].coutP, Ls[i].kp, Ls[i].Wh, Ls[i].sc, Ls[i].sh, Ls[i].hasBN);
    }

    auto gemm = [&](const _Float16* A, int ldA, int aOff, int li,
                    _Float16* oh, float* of, int ldOut, int outOff, int rows, float slope) {
        int coutP = Ls[li].coutP;
        int NT = (coutP >= 128) ? 8 : (coutP == 64 ? 4 : 2);
        int tilesM8 = rows / 128;                // 8 M-tiles per block
        int groupsN = coutP / (16 * NT);
        int blocks = tilesM8 * groupsN;          // all blocks full: no tail
        if (NT == 8)
            gemm_wmma_fused<8><<<blocks, 256, 0, stream>>>(
                A, ldA, aOff, Ls[li].Wh, Ls[li].kp, Ls[li].sc, Ls[li].sh,
                oh, of, ldOut, outOff, tilesM8, slope);
        else if (NT == 4)
            gemm_wmma_fused<4><<<blocks, 256, 0, stream>>>(
                A, ldA, aOff, Ls[li].Wh, Ls[li].kp, Ls[li].sc, Ls[li].sh,
                oh, of, ldOut, outOff, tilesM8, slope);
        else
            gemm_wmma_fused<2><<<blocks, 256, 0, stream>>>(
                A, ldA, aOff, Ls[li].Wh, Ls[li].kp, Ls[li].sc, Ls[li].sh,
                oh, of, ldOut, outOff, tilesM8, slope);
    };

    transpose_x<<<(2 * 3 * 8192 + 255) / 256, 256, 0, stream>>>(x, xt);
    knn3_kernel<<<NCHUNK, 256, 0, stream>>>(xt, idx);

    // ---- round 1
    gather_edge3<<<(int)(((long long)ROWS_E * 32 + 255) / 256), 256, 0, stream>>>(xt, idx, edge);
    gemm(edge, 32, 0, 0, oeA, nullptr, 128, 0, ROWS_E, 0.2f);          // c1
    gemm(oeA, 128, 0, 1, oeB, nullptr, 128, 0, ROWS_E, 0.2f);          // c2
    pool_maxmean<<<(NPTS * FD + 255) / 256, 256, 0, stream>>>(oeB, feat, 0, 128);
    gemm(feat, LDF, 0, 2, feat, nullptr, LDF, 256, NPTS, 0.2f);        // c3 -> o3

    // ---- round 2 (KNN in 128-D feature space via WMMA Gram)
    gram_wmma<<<(NCHUNK * 256) / 8, 256, 0, stream>>>(feat, 256, G);
    knn_feat_topk<<<NCHUNK, 256, 0, stream>>>(feat, 256, G, idx);
    gather_edge128<<<(int)(((long long)ROWS_E * FD + 255) / 256), 256, 0, stream>>>(feat, 256, idx, edge);
    gemm(edge, 256, 0, 3, oeA, nullptr, 128, 0, ROWS_E, 0.2f);         // c4
    pool_maxmean<<<(NPTS * FD + 255) / 256, 256, 0, stream>>>(oeA, feat, 384, 512);
    gemm(feat, LDF, 384, 4, feat, nullptr, LDF, 640, NPTS, 0.2f);      // c5 -> o5

    // ---- round 3
    gram_wmma<<<(NCHUNK * 256) / 8, 256, 0, stream>>>(feat, 640, G);
    knn_feat_topk<<<NCHUNK, 256, 0, stream>>>(feat, 640, G, idx);
    gather_edge128<<<(int)(((long long)ROWS_E * FD + 255) / 256), 256, 0, stream>>>(feat, 640, idx, edge);
    gemm(edge, 256, 0, 5, oeA, nullptr, 128, 0, ROWS_E, 0.2f);         // c6
    pool_maxmean<<<(NPTS * FD + 255) / 256, 256, 0, stream>>>(oeA, feat, 768, 896);
    gemm(feat, LDF, 768, 6, feat, nullptr, LDF, 1024, NPTS, 0.2f);     // c7 -> o7

    // ---- head MLP
    gemm(feat, LDF, 0, 7, h1b, nullptr, 256, 0, NPTS, 0.2f);           // h1
    gemm(h1b, 256, 0, 8, h2b, nullptr, 128, 0, NPTS, 0.2f);            // h2
    gemm(h2b, 128, 0, 9, h3b, nullptr, 64, 0, NPTS, 0.2f);             // h3
    gemm(h3b, 64, 0, 10, nullptr, coef, 32, 0, NPTS, 1.0f);            // h4 (linear, f32 out)

    finalize_kernel<<<(NPTS * 16 + 255) / 256, 256, 0, stream>>>(
        xt, coef, sel, uvset, (float*)d_out);
}
